// MSDeformAttn_56538949484672
// MI455X (gfx1250) — matrix operand order repaired and checked by reading, not compiled, not run on hardware
//
#include <hip/hip_runtime.h>
#include <hip/hip_bf16.h>

// ---------------------------------------------------------------------------
// MS-Deformable-Attention for MI455X (gfx1250, wave32, WMMA)
//   bs=2, nq=nv=22223, dm=256, NH=8, dh=32, nl=4, NP=4
// Pipeline:
//   [1] vproj  = value @ W_vp + b_vp                (WMMA bf16 GEMM, N=256)
//   [2] off    = query @ W_so + b_so                (WMMA bf16 GEMM, N=256)
//   [3] logits = query @ W_aw + b_aw                (WMMA bf16 GEMM, N=128)
//   [4] msout  = deformable sampling (in-wave softmax + bilinear gathers)
//   [5] out    = msout @ W_op + b_op + query        (WMMA bf16 GEMM + residual)
// ---------------------------------------------------------------------------

#define BS   2
#define NQ   22223
#define NV   22223
#define DM   256
#define NHD  8
#define DH   32

#define BM   64
#define BN   128
#define BK   32
#define LDA  40   // padded LDS stride (bf16 elems) for A tile
#define LDB  40   // padded LDS stride (bf16 elems) for B tile (transposed)

typedef __attribute__((ext_vector_type(8)))  __bf16 v8bf;
typedef __attribute__((ext_vector_type(16))) __bf16 v16bf;
typedef __attribute__((ext_vector_type(8)))  float  v8f;

__device__ __forceinline__ unsigned int f2bf(float f) {
    unsigned int u = __float_as_uint(f);
    u += 0x7FFFu + ((u >> 16) & 1u);   // round-to-nearest-even
    return u >> 16;
}
__device__ __forceinline__ unsigned int packbf2(float lo, float hi) {
    return f2bf(lo) | (f2bf(hi) << 16);
}

// ---------------------------------------------------------------------------
// GEMM: C[M,N] = A[M,256] * W[256,N] + bias[N] (+ residual[M,N])
// 256 threads = 8 waves; block tile 64x128; each wave -> 32x32 via 2x2 WMMA.
// A fragment (16x32 bf16): lane holds row M=lane%16; elems 0-7 -> K=(lane/16)*8+0..7,
//                          elems 8-15 -> +16 (per ISA 16-bit A layout).
// B fragment (32x16 bf16): lane holds col N=lane%16; elems 0-7 -> K=(lane/16)*8+0..7, 8-15 -> +16 (as A).
// Staging uses packed 8B DS stores only (A: row-major; B: register-transposed 4x4).
// ---------------------------------------------------------------------------
__global__ __launch_bounds__(256)
void gemm_bias_res_wmma(const float* __restrict__ A, const float* __restrict__ Wm,
                        const float* __restrict__ bias, const float* __restrict__ Rres,
                        float* __restrict__ C, int M, int N)
{
    __shared__ unsigned short As[BM * LDA];   // 5120 B
    __shared__ unsigned short Bsh[BN * LDB];  // 10240 B

    const int tid  = threadIdx.x;
    const int lane = tid & 31;
    const int wave = tid >> 5;
    const int wm   = wave >> 2;        // 0..1  (M direction)
    const int wn   = wave & 3;         // 0..3  (N direction)
    const int lmod = lane & 15;
    const int lhi  = lane >> 4;
    const int block_m = blockIdx.x * BM;
    const int block_n = blockIdx.y * BN;

    v8f acc[2][2];
#pragma unroll
    for (int i = 0; i < 2; i++)
#pragma unroll
        for (int j = 0; j < 2; j++)
#pragma unroll
            for (int e = 0; e < 8; e++) acc[i][j][e] = 0.0f;

    // per-thread staging coordinates (loop-invariant)
    const int arow0 = tid >> 3,  ac4 = tid & 7;      // A: rows 0..31 (+32 on 2nd pass)
    const int bnc   = tid & 31,  bkc = tid >> 5;     // B: 4x4 micro-tile (nc, kc)

    for (int kk = 0; kk < DM; kk += BK) {
        __syncthreads();
        // ---- stage A tile: 64 rows x 32 K (fp32 -> bf16), packed 8B DS stores
#pragma unroll
        for (int i = 0; i < 2; i++) {
            int arow = arow0 + i * 32;
            int grow = block_m + arow;
            float4 v = make_float4(0.f, 0.f, 0.f, 0.f);
            if (grow < M)
                v = *(const float4*)(A + (size_t)grow * DM + kk + ac4 * 4);
            *(uint2*)&As[arow * LDA + ac4 * 4] =
                make_uint2(packbf2(v.x, v.y), packbf2(v.z, v.w));
        }
        // ---- stage B tile: 32 K x 128 N -> Bsh[n][k] via register 4x4 transpose
        {
            const float* wp = Wm + (size_t)(kk + bkc * 4) * N + block_n + bnc * 4;
            float4 r0 = *(const float4*)(wp);
            float4 r1 = *(const float4*)(wp + N);
            float4 r2 = *(const float4*)(wp + 2 * N);
            float4 r3 = *(const float4*)(wp + 3 * N);
            unsigned short* bb = &Bsh[(bnc * 4) * LDB + bkc * 4];
            *(uint2*)(bb + 0 * LDB) = make_uint2(packbf2(r0.x, r1.x), packbf2(r2.x, r3.x));
            *(uint2*)(bb + 1 * LDB) = make_uint2(packbf2(r0.y, r1.y), packbf2(r2.y, r3.y));
            *(uint2*)(bb + 2 * LDB) = make_uint2(packbf2(r0.z, r1.z), packbf2(r2.z, r3.z));
            *(uint2*)(bb + 3 * LDB) = make_uint2(packbf2(r0.w, r1.w), packbf2(r2.w, r3.w));
        }
        __syncthreads();

        // ---- fragments (two aligned 16B ds_load_b128 each)
        v16bf afrag[2], bfrag[2];
#pragma unroll
        for (int mt = 0; mt < 2; mt++) {
            int rl = wm * 32 + mt * 16 + lmod;
            v8bf lo = *(const v8bf*)&As[rl * LDA + lhi * 8];
            v8bf hi = *(const v8bf*)&As[rl * LDA + lhi * 8 + 16];
            afrag[mt] = __builtin_shufflevector(lo, hi, 0,1,2,3,4,5,6,7,8,9,10,11,12,13,14,15);
        }
#pragma unroll
        for (int nt = 0; nt < 2; nt++) {
            int cl = wn * 32 + nt * 16 + lmod;
            v8bf lo = *(const v8bf*)&Bsh[cl * LDB + lhi * 8];
            v8bf hi = *(const v8bf*)&Bsh[cl * LDB + lhi * 8 + 16];
            bfrag[nt] = __builtin_shufflevector(lo, hi, 0,1,2,3,4,5,6,7,8,9,10,11,12,13,14,15);
        }
#pragma unroll
        for (int mt = 0; mt < 2; mt++)
#pragma unroll
            for (int nt = 0; nt < 2; nt++)
                acc[mt][nt] = __builtin_amdgcn_wmma_f32_16x16x32_bf16(
                    false, afrag[mt], false, bfrag[nt],
                    (short)0, acc[mt][nt], false, false);
    }

    // ---- epilogue: C/D layout -> lane holds col lmod, rows r + lhi*8
#pragma unroll
    for (int mt = 0; mt < 2; mt++)
#pragma unroll
        for (int nt = 0; nt < 2; nt++) {
            int gcol = block_n + wn * 32 + nt * 16 + lmod;
            float bv = bias[gcol];
#pragma unroll
            for (int e = 0; e < 8; e++) {
                int grow = block_m + wm * 32 + mt * 16 + lhi * 8 + e;
                if (grow < M) {
                    float o = acc[mt][nt][e] + bv;
                    if (Rres) o += Rres[(size_t)grow * N + gcol];
                    C[(size_t)grow * N + gcol] = o;
                }
            }
        }
}

// ---------------------------------------------------------------------------
// Sampling kernel: one 256-thread block per (b,q); wave h owns head h.
// Lane d (0..31) owns channel h*32+d -> each corner gather is one coalesced
// 128B line. Softmax over the 16 (l,p) logits via wave32 shfl butterflies.
// Offsets (32 floats) and ref points (8 floats) for this (q,h) are loaded
// once coalesced (one b32 per lane) and broadcast via __shfl.
// ---------------------------------------------------------------------------
__global__ __launch_bounds__(256)
void msda_sample(const float* __restrict__ vproj, const float* __restrict__ off,
                 const float* __restrict__ logits, const float* __restrict__ ref,
                 float* __restrict__ outbuf)
{
    const int LH[4] = {100, 50, 25, 13};
    const int LW[4] = {167, 84, 42, 21};
    const int LS[4] = {0, 16700, 20900, 21950};

    const int q    = blockIdx.x;
    const int b    = blockIdx.y;
    const int h    = threadIdx.x >> 5;
    const int lane = threadIdx.x & 31;
    const int row  = b * NQ + q;

    // in-wave softmax over 16 logits (both 16-lane halves hold identical data)
    float lg = logits[(size_t)row * (NHD * 16) + h * 16 + (lane & 15)];
    float m = lg;
    m = fmaxf(m, __shfl_xor(m, 1, 32));
    m = fmaxf(m, __shfl_xor(m, 2, 32));
    m = fmaxf(m, __shfl_xor(m, 4, 32));
    m = fmaxf(m, __shfl_xor(m, 8, 32));
    float e = __expf(lg - m);
    float s = e;
    s += __shfl_xor(s, 1, 32);
    s += __shfl_xor(s, 2, 32);
    s += __shfl_xor(s, 4, 32);
    s += __shfl_xor(s, 8, 32);
    float aw = e / s;

    // coalesced parameter loads: 32 offset floats, 8 ref floats per (q,h)
    float offv = off[(size_t)row * DM + h * 32 + lane];
    float refv = ref[(size_t)row * 8 + (lane & 7)];

    float acc = 0.f;
#pragma unroll
    for (int l = 0; l < 4; l++) {
        const int   Wl = LW[l], Hl = LH[l];
        const float rx = __shfl(refv, l * 2 + 0, 32);
        const float ry = __shfl(refv, l * 2 + 1, 32);
        const float* vbase = vproj + ((size_t)(b * NV + LS[l])) * DM + h * DH + lane;
#pragma unroll
        for (int p = 0; p < 4; p++) {
            const int lp = l * 4 + p;
            float a  = __shfl(aw,   lp,         32);  // softmax weight
            float ox = __shfl(offv, lp * 2 + 0, 32);
            float oy = __shfl(offv, lp * 2 + 1, 32);
            // loc = ref + off/(W,H); x = loc_x*W - 0.5 == rx*W + ox - 0.5
            float x = rx * (float)Wl + ox - 0.5f;
            float y = ry * (float)Hl + oy - 0.5f;
            float xf = floorf(x), yf = floorf(y);
            float lx = x - xf, ly = y - yf;
            int x0 = (int)xf, y0 = (int)yf;
            float w00 = (1.f - lx) * (1.f - ly) * a;
            float w10 = lx * (1.f - ly) * a;
            float w01 = (1.f - lx) * ly * a;
            float w11 = lx * ly * a;
            // corners (wave-uniform coords -> no divergence; 128B coalesced loads)
            if (x0 >= 0 && x0 < Wl && y0 >= 0 && y0 < Hl)
                acc += w00 * vbase[(size_t)(y0 * Wl + x0) * DM];
            if (x0 + 1 >= 0 && x0 + 1 < Wl && y0 >= 0 && y0 < Hl)
                acc += w10 * vbase[(size_t)(y0 * Wl + x0 + 1) * DM];
            if (x0 >= 0 && x0 < Wl && y0 + 1 >= 0 && y0 + 1 < Hl)
                acc += w01 * vbase[(size_t)((y0 + 1) * Wl + x0) * DM];
            if (x0 + 1 >= 0 && x0 + 1 < Wl && y0 + 1 >= 0 && y0 + 1 < Hl)
                acc += w11 * vbase[(size_t)((y0 + 1) * Wl + x0 + 1) * DM];
        }
    }
    outbuf[(size_t)row * DM + h * DH + lane] = acc;
}

// ---------------------------------------------------------------------------
extern "C" void kernel_launch(void* const* d_in, const int* in_sizes, int n_in,
                              void* d_out, int out_size, void* d_ws, size_t ws_size,
                              hipStream_t stream)
{
    const float* query  = (const float*)d_in[0];
    const float* value  = (const float*)d_in[1];
    const float* refpts = (const float*)d_in[2];
    const float* W_so   = (const float*)d_in[3];
    const float* b_so   = (const float*)d_in[4];
    const float* W_aw   = (const float*)d_in[5];
    const float* b_aw   = (const float*)d_in[6];
    const float* W_vp   = (const float*)d_in[7];
    const float* b_vp   = (const float*)d_in[8];
    const float* W_op   = (const float*)d_in[9];
    const float* b_op   = (const float*)d_in[10];
    float* out = (float*)d_out;

    const int M = BS * NQ;  // 44446

    float* vproj  = (float*)d_ws;                       // M*256
    float* offb   = vproj + (size_t)M * DM;             // M*256
    float* logit  = offb  + (size_t)M * DM;             // M*128
    float* msout  = logit + (size_t)M * 128;            // M*256

    dim3 blk(256);
    dim3 g256((M + BM - 1) / BM, DM / BN);
    dim3 g128((M + BM - 1) / BM, 128 / BN);

    gemm_bias_res_wmma<<<g256, blk, 0, stream>>>(value, W_vp, b_vp, nullptr, vproj, M, DM);
    gemm_bias_res_wmma<<<g256, blk, 0, stream>>>(query, W_so, b_so, nullptr, offb,  M, DM);
    gemm_bias_res_wmma<<<g128, blk, 0, stream>>>(query, W_aw, b_aw, nullptr, logit, M, 128);
    msda_sample<<<dim3(NQ, BS), blk, 0, stream>>>(vproj, offb, logit, refpts, msout);
    gemm_bias_res_wmma<<<g256, blk, 0, stream>>>(msout, W_op, b_op, query, out, M, DM);

    (void)in_sizes; (void)n_in; (void)out_size; (void)ws_size;
}
